// BiLSTM_CRF_56719338110979
// MI455X (gfx1250) — matrix operand order repaired and checked by reading, not compiled, not run on hardware
//
#include <hip/hip_runtime.h>
#include <hip/hip_bf16.h>

// ---------------------------------------------------------------------------
// BiLSTM-CRF Viterbi decode for MI455X (gfx1250)
//   Phase 0: one-time fp32 -> bf16 hi/lo split of W (removes 512x redundant
//            conversion VALU from the GEMM hot loop)
//   Phase 1: emissions GEMM (8192x2048 @ 2048x256) via bf16x3-split WMMA
//   Phase 2: sequential Viterbi scan on ONE WGP, transitions resident in LDS
//   Phase 3: LDS-staged backtrack
// ---------------------------------------------------------------------------

#define S_LEN 8192
#define HID   2048
#define TAGS  256
#define START_TAG 254
#define STOP_TAG  255
#define NEG_VAL  (-10000.0f)
#define TROW 260   // padded LDS row stride (floats): rows 16B-aligned,
                   // 260 mod 64 == 4 -> wave-consecutive rows spread banks

typedef __attribute__((ext_vector_type(16))) __bf16 v16bf;
typedef __attribute__((ext_vector_type(8)))  float  v8f;

static __device__ __forceinline__ __bf16 f2bf_trunc(float x) {
  union { float f; unsigned u; } c; c.f = x;
  union { unsigned short s; __bf16 b; } r;
  r.s = (unsigned short)(c.u >> 16);
  return r.b;
}
static __device__ __forceinline__ float bf2f(__bf16 b) {
  union { __bf16 b; unsigned short s; } c; c.b = b;
  union { unsigned u; float f; } r;
  r.u = ((unsigned)c.s) << 16;
  return r.f;
}

// ---------------------------------------------------------------------------
// Phase 0: split W (fp32 [256,2048]) into truncated-bf16 hi plane + residual
// lo plane, packed two elements per dword. Row-major, so GEMM B fragments are
// contiguous 16-element bf16 runs.
// ---------------------------------------------------------------------------
__global__ __launch_bounds__(256) void split_w(
    const float* __restrict__ W, unsigned* __restrict__ whi,
    unsigned* __restrict__ wlo) {
  const int idx = blockIdx.x * 256 + threadIdx.x;   // 0..262143 (pairs)
  float2 w = *(const float2*)(W + (size_t)idx * 2);
  const unsigned u0 = __float_as_uint(w.x);
  const unsigned u1 = __float_as_uint(w.y);
  const unsigned h0 = u0 & 0xFFFF0000u;
  const unsigned h1 = u1 & 0xFFFF0000u;
  const float l0 = w.x - __uint_as_float(h0);
  const float l1 = w.y - __uint_as_float(h1);
  whi[idx] = (u0 >> 16) | h1;
  wlo[idx] = (__float_as_uint(l0) >> 16) | (__float_as_uint(l1) & 0xFFFF0000u);
}

// ---------------------------------------------------------------------------
// Phase 1: feats[s,t] = sum_h hidden[s,h]*W[t,h] + b[t]
// One wave computes a 16-row x 64-col stripe (4 WMMA tiles), bf16x3 split:
//   A*B ~= Ahi*Bhi + Ahi*Blo + Alo*Bhi  (fp32 accumulate)
// A-frag (16x32 bf16): lane m=lane&15, lanes<16 hold K=0..15, lanes>=16 16..31
// B-frag (32x16 bf16): lane n=lane&15, lanes<16 hold K=0..15, lanes>=16 16..31
// ---------------------------------------------------------------------------
__global__ __launch_bounds__(256) void emissions_wmma(
    const float* __restrict__ hidden, const __bf16* __restrict__ whi,
    const __bf16* __restrict__ wlo, const float* __restrict__ bias,
    float* __restrict__ feats) {
  const int lane = threadIdx.x & 31;
  const int wid  = blockIdx.x * 8 + (threadIdx.x >> 5);   // 0..2047
  const int row_tile = wid >> 2;                          // 0..511
  const int cg       = wid & 3;                           // 0..3 (64-col group)
  const int row0 = row_tile * 16;
  const int col0 = cg * 64;
  const int m    = lane & 15;
  const int aoff = (lane < 16) ? 0 : 16;   // A K start
  const int boff = (lane < 16) ? 0 : 16;   // B K start

  v8f acc[4] = {};

  const float* arow = hidden + (size_t)(row0 + m) * HID;

  for (int kb = 0; kb < HID; kb += 32) {
    // ---- A fragment: one 16-float run (K = kb+aoff .. kb+aoff+15)
    float4 a0 = *(const float4*)(arow + kb + aoff);
    float4 a1 = *(const float4*)(arow + kb + aoff + 4);
    float4 a2 = *(const float4*)(arow + kb + 8 + aoff);
    float4 a3 = *(const float4*)(arow + kb + 8 + aoff + 4);
    float af[16];
    af[0]=a0.x; af[1]=a0.y; af[2]=a0.z; af[3]=a0.w;
    af[4]=a1.x; af[5]=a1.y; af[6]=a1.z; af[7]=a1.w;
    af[8]=a2.x; af[9]=a2.y; af[10]=a2.z; af[11]=a2.w;
    af[12]=a3.x; af[13]=a3.y; af[14]=a3.z; af[15]=a3.w;
    v16bf ahi, alo;
#pragma unroll
    for (int i = 0; i < 16; ++i) {
      __bf16 h = f2bf_trunc(af[i]);
      ahi[i] = h;
      alo[i] = f2bf_trunc(af[i] - bf2f(h));
    }

#pragma unroll
    for (int nt = 0; nt < 4; ++nt) {
      // ---- B fragments: pre-split bf16, one contiguous 32B run each
      const size_t boffs = (size_t)(col0 + nt * 16 + m) * HID + kb + boff;
      v16bf bhi = *(const v16bf*)(whi + boffs);
      v16bf blo = *(const v16bf*)(wlo + boffs);
      acc[nt] = __builtin_amdgcn_wmma_f32_16x16x32_bf16(
          false, ahi, false, bhi, (short)0, acc[nt], false, false);
      acc[nt] = __builtin_amdgcn_wmma_f32_16x16x32_bf16(
          false, ahi, false, blo, (short)0, acc[nt], false, false);
      acc[nt] = __builtin_amdgcn_wmma_f32_16x16x32_bf16(
          false, alo, false, bhi, (short)0, acc[nt], false, false);
    }
  }

  // ---- epilogue: C layout = lanes 0-15 N=lane,M=j; lanes 16-31 N=lane-16,M=j+8
  const int n_l   = lane & 15;
  const int mbase = (lane >> 4) * 8;
#pragma unroll
  for (int nt = 0; nt < 4; ++nt) {
    const int col = col0 + nt * 16 + n_l;
    const float bb = bias[col];
#pragma unroll
    for (int j = 0; j < 8; ++j) {
      feats[(size_t)(row0 + mbase + j) * TAGS + col] = acc[nt][j] + bb;
    }
  }
}

// ---------------------------------------------------------------------------
// Phase 2+3: Viterbi scan (single workgroup = 32 wave32s on one WGP).
// transitions[next][prev] lives in LDS (256x260 padded rows, ~260 KB of the
// 320 KB WGP LDS). Thread t handles next = t&255, prev quarter q = t>>8.
// Strict '>' with ascending prev / ascending quarter reproduces jnp.argmax
// first-occurrence tie-breaking. feats[s] is prefetched at the top of each
// step so its L2 latency overlaps the max-plus compute instead of sitting on
// the serial chain. Backpointers (1 byte each, T=256) go to d_ws; backtrack
// re-stages them through LDS in 1024-step chunks.
// ---------------------------------------------------------------------------
__global__ __launch_bounds__(1024) void viterbi_scan(
    const float* __restrict__ trans, const float* __restrict__ feats,
    unsigned char* __restrict__ bp, float* __restrict__ out) {
  __shared__ float transP[TAGS * TROW];   // 266,240 B
  __shared__ float fv[TAGS];
  __shared__ float red[4 * TAGS];
  __shared__ int   redi[4 * TAGS];

  const int tid = threadIdx.x;

  // load transitions (coalesced global, conflict-free LDS: consecutive prev)
  for (int idx = tid; idx < TAGS * TAGS; idx += 1024) {
    const int nn = idx >> 8, pp = idx & 255;
    transP[nn * TROW + pp] = trans[idx];
  }
  if (tid < TAGS) fv[tid] = (tid == START_TAG) ? 0.0f : NEG_VAL;
  __syncthreads();

  const int n  = tid & 255;
  const int q  = tid >> 8;
  const int p0 = q * 64;
  const float* trow = transP + n * TROW + p0;
  const float* fq   = fv + p0;

  for (int s = 0; s < S_LEN; ++s) {
    // prefetch this step's emission early (independent of fv)
    float feat_s = 0.0f;
    if (tid < TAGS) feat_s = feats[(size_t)s * TAGS + n];

    float best = -3.0e38f;
    int   bidx = p0;
#pragma unroll
    for (int i = 0; i < 64; i += 4) {
      float4 tv  = *(const float4*)(trow + i);
      float4 fvv = *(const float4*)(fq + i);
      float v0 = tv.x + fvv.x;
      float v1 = tv.y + fvv.y;
      float v2 = tv.z + fvv.z;
      float v3 = tv.w + fvv.w;
      if (v0 > best) { best = v0; bidx = p0 + i + 0; }
      if (v1 > best) { best = v1; bidx = p0 + i + 1; }
      if (v2 > best) { best = v2; bidx = p0 + i + 2; }
      if (v3 > best) { best = v3; bidx = p0 + i + 3; }
    }
    red[q * TAGS + n]  = best;
    redi[q * TAGS + n] = bidx;
    __syncthreads();
    if (tid < TAGS) {
      float m0 = red[n];
      int   bi = redi[n];
#pragma unroll
      for (int qq = 1; qq < 4; ++qq) {
        const float v = red[qq * TAGS + n];
        if (v > m0) { m0 = v; bi = redi[qq * TAGS + n]; }
      }
      fv[n] = m0 + feat_s;
      bp[(size_t)s * TAGS + n] = (unsigned char)bi;
    }
    __syncthreads();
  }

  // terminal = fv + transitions[STOP, :]
  if (tid < TAGS) red[tid] = fv[tid] + transP[STOP_TAG * TROW + tid];
  __syncthreads();

  int best_tag = 0;
  if (tid == 0) {
    float m0 = red[0];
    for (int t = 1; t < TAGS; ++t) {
      if (red[t] > m0) { m0 = red[t]; best_tag = t; }
    }
    out[0] = m0;
  }
  __syncthreads();  // transitions region is dead beyond this point

  // backtrack: stage 1024-step chunks (256 KB) of backpointers into LDS
  unsigned char* stage = (unsigned char*)transP;
  for (int chunk = 7; chunk >= 0; --chunk) {
    const unsigned* src = (const unsigned*)(bp + (size_t)chunk * 1024 * TAGS);
    unsigned* dst = (unsigned*)stage;
    for (int idx = tid; idx < 65536; idx += 1024) dst[idx] = src[idx];
    __syncthreads();
    if (tid == 0) {
      const int base = chunk * 1024;
      for (int s = 1023; s >= 0; --s) {
        out[1 + base + s] = (float)best_tag;
        best_tag = stage[s * TAGS + best_tag];
      }
    }
    __syncthreads();
  }
}

// ---------------------------------------------------------------------------
extern "C" void kernel_launch(void* const* d_in, const int* in_sizes, int n_in,
                              void* d_out, int out_size, void* d_ws, size_t ws_size,
                              hipStream_t stream) {
  const float* hidden = (const float*)d_in[0];  // [1, 8192, 2048]
  const float* W      = (const float*)d_in[1];  // [256, 2048]
  const float* bias   = (const float*)d_in[2];  // [256]
  const float* trans  = (const float*)d_in[3];  // [256, 256]
  float* out = (float*)d_out;                   // [1 + 8192]

  // workspace layout: feats 8MB | bp 2MB | Whi 1MB | Wlo 1MB  (12 MB total)
  float* feats = (float*)d_ws;
  unsigned char* bp =
      (unsigned char*)d_ws + (size_t)S_LEN * TAGS * sizeof(float);
  unsigned* whi = (unsigned*)(bp + (size_t)S_LEN * TAGS);
  unsigned* wlo = whi + (TAGS * HID / 2);

  // Phase 0: split W into bf16 hi/lo planes (262144 dword-pairs)
  split_w<<<1024, 256, 0, stream>>>(W, whi, wlo);
  // Phase 1: 2048 waves = 512 row-tiles x 4 col-groups, 8 waves per block
  emissions_wmma<<<256, 256, 0, stream>>>(hidden, (const __bf16*)whi,
                                          (const __bf16*)wlo, bias, feats);
  // Phase 2+3: single-WGP sequential scan + backtrack
  viterbi_scan<<<1, 1024, 0, stream>>>(trans, feats, bp, out);
}
